// TorchLMHeadJSD_17102559773043
// MI455X (gfx1250) — matrix operand order repaired and checked by reading, not compiled, not run on hardware
//
#include <hip/hip_runtime.h>
#include <hip/hip_bf16.h>
#include <math.h>

// ---------------------------------------------------------------------------
// LM-head JSD on MI455X (gfx1250, wave32).
//
// Pass 1: per-token logsumexp of student/teacher logits (streaming GEMM +
//         online softmax; 25 V-splits write partial (max,sum) pairs).
// Combine: merge partials -> Zs[2048], Zt[2048].
// Pass 2: recompute logit tiles, accumulate beta*p*lp+(1-b)*q*lq - m*log m.
// Finalize: mask by label != -100, mean.
//
// GEMMs: split-bf16 (hi+lo) 3x V_WMMA_F32_16X16X32_BF16 for ~fp32 accuracy.
// Tiles: 128 tokens x 256 vocab per 256-thread workgroup (8 waves), each wave
// owns TWO 16-column strips (A-fragment loads feed 6 WMMAs instead of 3:
// ~16 LDS bytes/WMMA). K chunks of 64 (2 WMMA K-steps) staged in LDS as bf16
// hi/lo planes; float4 global loads, 4xbf16 packed ds_store_b64.
// LDS: 96 KB tiles + ~9 KB reductions (<320 KB CDNA5 workgroup limit).
// Workspace use: 4*25*2048 + 3*2048 floats (~844 KB).
// ---------------------------------------------------------------------------

#define BT      2048
#define VV      32000
#define H_S     2048
#define H_T     4096
#define IGNORE  (-100)
#define NSPLIT  25          // V splits: 25 * 1280 = 32000
#define COLS_PER_SPLIT 1280 // 5 iterations of 256 columns
#define NITER   5
#define MBLK    128         // tokens per workgroup
#define NBLK    256         // vocab columns per iteration (8 waves * 2 * 16)
#define KC      64          // K chunk staged in LDS (2 WMMA K-steps)

typedef __attribute__((ext_vector_type(16))) __bf16 v16bf;
typedef __attribute__((ext_vector_type(8)))  float  v8f;

union FragU {
    v16bf v;
    uint4 q[2];
};
union F4 {
    float4 v;
    float  f[4];
};
union Pack4 {
    __bf16 h[4];
    uint2  u;
};

#define WMMA_BF16(A, B, C) \
    __builtin_amdgcn_wmma_f32_16x16x32_bf16(false, (A), false, (B), (short)0, (C), false, false)

// Compute 2x8 16x16 f32 accumulators per wave: rows [rowBase, rowBase+128) x
// cols [colBase + c*128 + wave*16, +16) for strips c=0,1.
// X:[BT][H], W:[V][H] row-major.  Uniform control flow (EXEC all-ones).
__device__ __forceinline__ void gemm_tile(
    const float* __restrict__ X, const float* __restrict__ W, int H,
    int rowBase, int colBase,
    __bf16* Ah, __bf16* Al, __bf16* Bh, __bf16* Bl,
    v8f (&acc)[2][8])
{
    const int tid  = threadIdx.x;
    const int wave = tid >> 5;
    const int lane = tid & 31;
    const int g    = lane >> 4;   // 16-lane half (ISA 7.12.2 layouts)
    const int n16  = lane & 15;

#pragma unroll
    for (int c = 0; c < 2; ++c)
#pragma unroll
        for (int s = 0; s < 8; ++s)
#pragma unroll
            for (int r = 0; r < 8; ++r) acc[c][s][r] = 0.0f;

    for (int k0 = 0; k0 < H; k0 += KC) {
        __syncthreads();   // previous chunk's fragment reads done
        // Stage A (activations, 128 rows) as bf16 hi/lo planes.
        for (int vid = tid; vid < MBLK * (KC / 4); vid += 256) {
            const int r  = vid >> 4;          // KC/4 = 16 float4 per row
            const int k4 = (vid & 15) << 2;
            F4 x;
            x.v = *(const float4*)(X + (size_t)(rowBase + r) * H + (k0 + k4));
            Pack4 xh, xl;
#pragma unroll
            for (int j = 0; j < 4; ++j) {
                __bf16 h = (__bf16)x.f[j];
                xh.h[j] = h;
                xl.h[j] = (__bf16)(x.f[j] - (float)h);
            }
            *(uint2*)&Ah[(size_t)r * KC + k4] = xh.u;
            *(uint2*)&Al[(size_t)r * KC + k4] = xl.u;
        }
        // Stage B (weights, 256 rows) as bf16 hi/lo planes.
        for (int vid = tid; vid < NBLK * (KC / 4); vid += 256) {
            const int r  = vid >> 4;
            const int k4 = (vid & 15) << 2;
            F4 w;
            w.v = *(const float4*)(W + (size_t)(colBase + r) * H + (k0 + k4));
            Pack4 wh, wl;
#pragma unroll
            for (int j = 0; j < 4; ++j) {
                __bf16 h = (__bf16)w.f[j];
                wh.h[j] = h;
                wl.h[j] = (__bf16)(w.f[j] - (float)h);
            }
            *(uint2*)&Bh[(size_t)r * KC + k4] = wh.u;
            *(uint2*)&Bl[(size_t)r * KC + k4] = wl.u;
            if (k0 + KC < H)  // global_prefetch_b8 for next weight chunk
                __builtin_prefetch(&W[(size_t)(colBase + r) * H + (k0 + KC + k4)], 0, 1);
        }
        __syncthreads();

#pragma unroll
        for (int kk = 0; kk < KC; kk += 32) {
            // B fragments (32x16 bf16): K = kk+g*16+e, col strip c.
            FragU bh[2], bl[2];
#pragma unroll
            for (int c = 0; c < 2; ++c) {
                const __bf16* bp = Bh + (size_t)((c << 7) + (wave << 4) + n16) * KC + kk + (g << 4);
                bh[c].q[0] = *(const uint4*)bp;
                bh[c].q[1] = *(const uint4*)(bp + 8);
                const __bf16* blp = Bl + (size_t)((c << 7) + (wave << 4) + n16) * KC + kk + (g << 4);
                bl[c].q[0] = *(const uint4*)blp;
                bl[c].q[1] = *(const uint4*)(blp + 8);
            }
            // A fragments (hi+lo, 8 sub-tiles); each feeds 6 WMMAs (2 strips).
            FragU ah[8], al[8];
#pragma unroll
            for (int s = 0; s < 8; ++s) {
                // A (16x32 bf16): row = s*16+n16; K = kk + g*16+e
                const __bf16* ap = Ah + (size_t)((s << 4) + n16) * KC + kk;
                ah[s].q[0] = *(const uint4*)(ap + (g << 4));
                ah[s].q[1] = *(const uint4*)(ap + 8 + (g << 4));
                const __bf16* alp = Al + (size_t)((s << 4) + n16) * KC + kk;
                al[s].q[0] = *(const uint4*)(alp + (g << 4));
                al[s].q[1] = *(const uint4*)(alp + 8 + (g << 4));
            }
#pragma unroll
            for (int s = 0; s < 8; ++s) {
#pragma unroll
                for (int c = 0; c < 2; ++c) {
                    // split-bf16 fp32 emulation: hi*hi + hi*lo + lo*hi
                    acc[c][s] = WMMA_BF16(ah[s].v, bh[c].v, acc[c][s]);
                    acc[c][s] = WMMA_BF16(ah[s].v, bl[c].v, acc[c][s]);
                    acc[c][s] = WMMA_BF16(al[s].v, bh[c].v, acc[c][s]);
                }
            }
        }
    }
}

// ---------------------------------------------------------------------------
// Pass 1: partial (rowmax, sumexp) over this block's V slice.
// grid (BT/128, NSPLIT), block 256.
// ---------------------------------------------------------------------------
__global__ __launch_bounds__(256, 2)
void lse_partial_kernel(const float* __restrict__ X,
                        const float* __restrict__ W, int H,
                        float* __restrict__ pMax,
                        float* __restrict__ pSum)
{
    __shared__ __attribute__((aligned(16))) __bf16 Ah[MBLK * KC];
    __shared__ __attribute__((aligned(16))) __bf16 Al[MBLK * KC];
    __shared__ __attribute__((aligned(16))) __bf16 Bh[NBLK * KC];
    __shared__ __attribute__((aligned(16))) __bf16 Bl[NBLK * KC];
    __shared__ float redMax[8][MBLK];
    __shared__ float redSum[8][MBLK];
    __shared__ float runMax[MBLK];
    __shared__ float runSum[MBLK];

    const int tid  = threadIdx.x;
    const int wave = tid >> 5;
    const int lane = tid & 31;
    const int g    = lane >> 4;
    const int rowBase = blockIdx.x * MBLK;

    if (tid < MBLK) { runMax[tid] = -INFINITY; runSum[tid] = 0.0f; }

    for (int it = 0; it < NITER; ++it) {
        const int colBase = blockIdx.y * COLS_PER_SPLIT + it * NBLK;
        v8f acc[2][8];
        gemm_tile(X, W, H, rowBase, colBase, Ah, Al, Bh, Bl, acc);

        // Row-wise max and sumexp over this wave's 32 columns (2 strips).
        // C layout: lane l, vgpr r -> row = 16s + r + 8*(l/16), col = l%16.
#pragma unroll
        for (int s = 0; s < 8; ++s) {
#pragma unroll
            for (int r = 0; r < 8; ++r) {
                const float v0 = acc[0][s][r];
                const float v1 = acc[1][s][r];
                float mx = fmaxf(v0, v1);
                for (int o = 8; o; o >>= 1) mx = fmaxf(mx, __shfl_xor(mx, o));
                float e = __expf(v0 - mx) + __expf(v1 - mx);
                for (int o = 8; o; o >>= 1) e += __shfl_xor(e, o);
                if ((lane & 15) == 0) {
                    const int row = (s << 4) + (g << 3) + r;
                    redMax[wave][row] = mx;
                    redSum[wave][row] = e;
                }
            }
        }
        __syncthreads();
        if (tid < MBLK) {  // merge 8 wave partials into running (max,sum)
            float m = runMax[tid], ssum = runSum[tid];
#pragma unroll
            for (int w = 0; w < 8; ++w) {
                const float wm  = redMax[w][tid];
                const float wsv = redSum[w][tid];
                const float nm  = fmaxf(m, wm);
                ssum = ssum * __expf(m - nm) + wsv * __expf(wm - nm);
                m = nm;
            }
            runMax[tid] = m;
            runSum[tid] = ssum;
        }
        // next gemm_tile() begins with __syncthreads(): no hazard on red*/run*
    }
    __syncthreads();
    if (tid < MBLK) {
        const int token = rowBase + tid;
        pMax[blockIdx.y * BT + token] = runMax[tid];
        pSum[blockIdx.y * BT + token] = runSum[tid];
    }
}

// ---------------------------------------------------------------------------
// Combine NSPLIT partials -> Z = max + log(sumexp).  4096 threads total.
// ---------------------------------------------------------------------------
__global__ void lse_combine_kernel(const float* __restrict__ sMax,
                                   const float* __restrict__ sSum,
                                   const float* __restrict__ tMax,
                                   const float* __restrict__ tSum,
                                   float* __restrict__ Zs,
                                   float* __restrict__ Zt)
{
    const int id = blockIdx.x * blockDim.x + threadIdx.x;
    if (id >= 2 * BT) return;
    const bool stu = (id < BT);
    const float* pM = stu ? sMax : tMax;
    const float* pS = stu ? sSum : tSum;
    float* Z        = stu ? Zs : Zt;
    const int token = id & (BT - 1);

    float m = -INFINITY;
    for (int p = 0; p < NSPLIT; ++p) m = fmaxf(m, pM[p * BT + token]);
    float s = 0.0f;
    for (int p = 0; p < NSPLIT; ++p)
        s += pS[p * BT + token] * __expf(pM[p * BT + token] - m);
    Z[token] = m + __logf(s);
}

// ---------------------------------------------------------------------------
// Pass 2: recompute both logit tiles and accumulate the JSD sum.
// per_token = 0.5*Sum p*lp + 0.5*Sum q*lq - Sum m*log m,  m = 0.5(p+q).
// grid (BT/128, NSPLIT), block 256.
// ---------------------------------------------------------------------------
__global__ __launch_bounds__(256, 2)
void jsd_kernel(const float* __restrict__ Xs,
                const float* __restrict__ Ws,
                const float* __restrict__ Xt,
                const float* __restrict__ Wt,
                const float* __restrict__ Zs,
                const float* __restrict__ Zt,
                float* __restrict__ lossAcc)
{
    __shared__ __attribute__((aligned(16))) __bf16 Ah[MBLK * KC];
    __shared__ __attribute__((aligned(16))) __bf16 Al[MBLK * KC];
    __shared__ __attribute__((aligned(16))) __bf16 Bh[NBLK * KC];
    __shared__ __attribute__((aligned(16))) __bf16 Bl[NBLK * KC];
    __shared__ float redSum[8][MBLK];
    __shared__ float runLoss[MBLK];
    __shared__ float ZsS[MBLK];
    __shared__ float ZtS[MBLK];

    const int tid  = threadIdx.x;
    const int wave = tid >> 5;
    const int lane = tid & 31;
    const int g    = lane >> 4;
    const int rowBase = blockIdx.x * MBLK;

    if (tid < MBLK) {
        runLoss[tid] = 0.0f;
        ZsS[tid] = Zs[rowBase + tid];
        ZtS[tid] = Zt[rowBase + tid];
    }

    for (int it = 0; it < NITER; ++it) {
        const int colBase = blockIdx.y * COLS_PER_SPLIT + it * NBLK;
        v8f accS[2][8], accT[2][8];
        gemm_tile(Xs, Ws, H_S, rowBase, colBase, Ah, Al, Bh, Bl, accS);
        gemm_tile(Xt, Wt, H_T, rowBase, colBase, Ah, Al, Bh, Bl, accT);

#pragma unroll
        for (int s = 0; s < 8; ++s) {
#pragma unroll
            for (int r = 0; r < 8; ++r) {
                const int row = (s << 4) + (g << 3) + r;
                const float zs = ZsS[row];
                const float zt = ZtS[row];
                float t = 0.0f;
#pragma unroll
                for (int c = 0; c < 2; ++c) {
                    const float ls = accS[c][s][r] - zs;  // log q (student)
                    const float lt = accT[c][s][r] - zt;  // log p (teacher)
                    const float q  = __expf(ls);
                    const float p  = __expf(lt);
                    const float m  = 0.5f * (p + q);
                    float tc = 0.5f * (p * lt + q * ls);
                    if (m > 0.0f) tc -= m * __logf(m);
                    t += tc;
                }
                // sum over 16 columns (16-lane halves keep rows separate)
                for (int o = 8; o; o >>= 1) t += __shfl_xor(t, o);
                if ((lane & 15) == 0) redSum[wave][row] = t;
            }
        }
        __syncthreads();
        if (tid < MBLK) {
            float tot = 0.0f;
#pragma unroll
            for (int w = 0; w < 8; ++w) tot += redSum[w][tid];
            runLoss[tid] += tot;
        }
    }
    __syncthreads();
    if (tid < MBLK) atomicAdd(&lossAcc[rowBase + tid], runLoss[tid]);
}

// ---------------------------------------------------------------------------
__global__ void zero_kernel(float* __restrict__ p, int n)
{
    const int i = blockIdx.x * blockDim.x + threadIdx.x;
    if (i < n) p[i] = 0.0f;
}

__global__ void finalize_kernel(const float* __restrict__ lossAcc,
                                const long long* __restrict__ label,
                                float* __restrict__ out)
{
    __shared__ float ssum[256];
    __shared__ int   scnt[256];
    const int tid = threadIdx.x;
    float s = 0.0f;
    int   c = 0;
    for (int i = tid; i < BT; i += 256) {
        if (label[i] != (long long)IGNORE) { s += lossAcc[i]; ++c; }
    }
    ssum[tid] = s;
    scnt[tid] = c;
    __syncthreads();
    for (int off = 128; off; off >>= 1) {
        if (tid < off) { ssum[tid] += ssum[tid + off]; scnt[tid] += scnt[tid + off]; }
        __syncthreads();
    }
    if (tid == 0) out[0] = scnt[0] ? (ssum[0] / (float)scnt[0]) : 0.0f;
}

// ---------------------------------------------------------------------------
extern "C" void kernel_launch(void* const* d_in, const int* in_sizes, int n_in,
                              void* d_out, int out_size, void* d_ws, size_t ws_size,
                              hipStream_t stream)
{
    (void)in_sizes; (void)n_in; (void)out_size; (void)ws_size;
    const float*     Xs    = (const float*)d_in[0];      // [2048][2048]
    const float*     Xt    = (const float*)d_in[1];      // [2048][4096]
    const float*     Ws    = (const float*)d_in[2];      // [32000][2048]
    const float*     Wt    = (const float*)d_in[3];      // [32000][4096]
    const long long* label = (const long long*)d_in[4];  // [2048] int64
    float* out = (float*)d_out;

    float* ws      = (float*)d_ws;
    float* sMax    = ws;                        // [NSPLIT][BT]
    float* sSum    = sMax + NSPLIT * BT;
    float* tMax    = sSum + NSPLIT * BT;
    float* tSum    = tMax + NSPLIT * BT;
    float* Zs      = tSum + NSPLIT * BT;        // [BT]
    float* Zt      = Zs + BT;
    float* lossAcc = Zt + BT;                   // [BT]

    zero_kernel<<<(BT + 255) / 256, 256, 0, stream>>>(lossAcc, BT);

    dim3 grid(BT / MBLK, NSPLIT);
    lse_partial_kernel<<<grid, 256, 0, stream>>>(Xs, Ws, H_S, sMax, sSum);
    lse_partial_kernel<<<grid, 256, 0, stream>>>(Xt, Wt, H_T, tMax, tSum);
    lse_combine_kernel<<<(2 * BT + 255) / 256, 256, 0, stream>>>(
        sMax, sSum, tMax, tSum, Zs, Zt);
    jsd_kernel<<<grid, 256, 0, stream>>>(Xs, Ws, Xt, Wt, Zs, Zt, lossAcc);
    finalize_kernel<<<1, 256, 0, stream>>>(lossAcc, label, out);
}
